// sLSTM_29145648071333
// MI455X (gfx1250) — matrix operand order repaired and checked by reading, not compiled, not run on hardware
//
#include <hip/hip_runtime.h>

#define IN_DIM   64
#define HID      256
#define BATCH    128
#define TLEN     512
#define NWG      16
#define NTHREADS 256

#if defined(__AMDGCN__) && __has_builtin(__builtin_amdgcn_tensor_load_to_lds)
#define HAVE_TDM 1
#else
#define HAVE_TDM 0
#endif

typedef __attribute__((ext_vector_type(16))) __bf16       v16bf;
typedef __attribute__((ext_vector_type(8)))  float        v8f;
typedef __attribute__((ext_vector_type(4)))  unsigned int v4u;
typedef __attribute__((ext_vector_type(4)))  unsigned int u32x4;
typedef __attribute__((ext_vector_type(8)))  int          i32x8;
typedef __attribute__((ext_vector_type(4)))  int          i32x4;

union FragU { v4u u[2]; v16bf v; };

__device__ inline unsigned short f2bf(float f) {
  unsigned int u = __float_as_uint(f);
  u += 0x7FFFu + ((u >> 16) & 1u);          // round-to-nearest-even
  return (unsigned short)(u >> 16);
}

__device__ inline v8f vzero() {
  v8f v;
#pragma unroll
  for (int e = 0; e < 8; ++e) v[e] = 0.f;
  return v;
}

// A-matrix fragment (16x32 bf16, row-major source, per ISA 7.12.2):
// lane L (L<16): row=L,    VGPR0..3 hold K=0..7,  VGPR4..7 hold K=16..23
// lane L (>=16): row=L-16, VGPR0..3 hold K=8..15, VGPR4..7 hold K=24..31
__device__ inline v16bf load_frag_a(const unsigned short* base, int stride, int lane) {
  const int r  = lane & 15;
  const int kb = (lane >> 4) << 3;          // 0 or 8 elements
  const unsigned short* p = base + r * stride + kb;
  FragU f;
  f.u[0] = *reinterpret_cast<const v4u*>(p);        // K = kb .. kb+7
  f.u[1] = *reinterpret_cast<const v4u*>(p + 16);   // K = kb+16 .. kb+23
  return f.v;
}

// B-matrix fragment (32x16 bf16). B[k][n] = W[n][k], so a row-major W slice
// gives column-contiguous B: lane L (col = L&15) needs K = kb..kb+7, kb+16..kb+23 (same as A).
__device__ inline v16bf load_frag_b(const unsigned short* base, int stride, int lane) {
  const int c  = lane & 15;
  const int kb = (lane >> 4) << 3;          // 0 or 8 elements
  const unsigned short* p = base + c * stride + kb;
  FragU f;
  f.u[0] = *reinterpret_cast<const v4u*>(p);
  f.u[1] = *reinterpret_cast<const v4u*>(p + 16);
  return f.v;
}

// Tensor Data Mover: 1-D DMA of one x timestep tile (128*64 bf16 = 16 KB,
// moved as 2048 x 8-byte units) from global into an LDS double buffer.
// D# per CDNA5 ISA: group0 = {count, lds_addr, global_addr,
// type=2}; group1 = {mask=0, data_size=3(8B), tensor_dim0=tile_dim0=2048}.
__device__ inline void tdm_load_x_tile(const unsigned short* gsrc, unsigned lds_addr) {
#if HAVE_TDM
  unsigned long long ga = (unsigned long long)gsrc;
  u32x4 g0;
  g0[0] = 1u;                                             // count=1 (valid user D#)
  g0[1] = lds_addr;                                       // LDS byte address
  g0[2] = (unsigned)(ga & 0xFFFFFFFFu);                   // global_addr[31:0]
  g0[3] = (unsigned)((ga >> 32) & 0x01FFFFFFu) | (2u << 30);  // [56:32] | type=2
  i32x8 g1;
  g1[0] = (3 << 16);              // workgroup_mask=0 (no cluster), data_size=3 (8B)
  g1[1] = (int)(2048u << 16);     // tensor_dim0[15:0] @ bits[63:48]
  g1[2] = 0;                      // tensor_dim0 hi / tensor_dim1 lo
  g1[3] = (int)(2048u << 16);     // tile_dim0 @ bits[127:112]
  g1[4] = 0;                      // tile_dim1=0 (1-D), tile_dim2=0
  g1[5] = 0; g1[6] = 0; g1[7] = 0;
  i32x4 gz = {0, 0, 0, 0};
#if __clang_major__ >= 23
  i32x8 gz8 = {0, 0, 0, 0, 0, 0, 0, 0};
  __builtin_amdgcn_tensor_load_to_lds(g0, g1, gz, gz, gz8, 0);
#else
  __builtin_amdgcn_tensor_load_to_lds(g0, g1, gz, gz, 0);
#endif
#else
  (void)gsrc; (void)lds_addr;
#endif
}

__device__ inline void tdm_wait_all() {
#if HAVE_TDM
  __builtin_amdgcn_s_wait_tensorcnt(0);
#endif
}

__device__ inline float sigmoid_fast(float x) { return 1.f / (1.f + __expf(-x)); }
__device__ inline float tanh_fast(float x) {
  float xc = fminf(fmaxf(x, -15.f), 15.f);
  float t  = __expf(2.f * xc);
  return (t - 1.f) / (t + 1.f);
}

__device__ inline void grid_sync(unsigned int* cnt, unsigned int& epoch) {
  __threadfence();
  __syncthreads();
  epoch += 1;
  if (threadIdx.x == 0) {
    __hip_atomic_fetch_add(cnt, 1u, __ATOMIC_ACQ_REL, __HIP_MEMORY_SCOPE_AGENT);
    while (__hip_atomic_load(cnt, __ATOMIC_ACQUIRE, __HIP_MEMORY_SCOPE_AGENT) <
           epoch * (unsigned)NWG) {
      __builtin_amdgcn_s_sleep(1);
    }
  }
  __syncthreads();
  __threadfence();
}

// Convert x [B,T,IN] fp32 -> xt [T,B,IN] bf16 so the time loop reads contiguous tiles.
__global__ void xt_kernel(const float* __restrict__ x, unsigned short* __restrict__ xt) {
  int i = blockIdx.x * 256 + threadIdx.x;
  if (i >= BATCH * TLEN * IN_DIM) return;
  int b = i / (TLEN * IN_DIM);
  int rem = i - b * (TLEN * IN_DIM);
  int t = rem / IN_DIM;
  int k = rem - t * IN_DIM;
  xt[(t * BATCH + b) * IN_DIM + k] = f2bf(x[i]);
}

__global__ __launch_bounds__(NTHREADS, 1)
void slstm_persistent(const unsigned short* __restrict__ xt,
                      unsigned short* __restrict__ h1a, unsigned short* __restrict__ h1b,
                      unsigned short* __restrict__ h2a, unsigned short* __restrict__ h2b,
                      float* __restrict__ h2f,
                      const float* __restrict__ Wxh1, const float* __restrict__ bxh1,
                      const float* __restrict__ Whh1, const float* __restrict__ bhh1,
                      const float* __restrict__ Wxh2, const float* __restrict__ bxh2,
                      const float* __restrict__ Whh2, const float* __restrict__ bhh2,
                      const float* __restrict__ W1, const float* __restrict__ b1,
                      const float* __restrict__ W2, const float* __restrict__ b2,
                      const float* __restrict__ W3, const float* __restrict__ b3,
                      float* __restrict__ out, unsigned int* __restrict__ cnt) {
  // Per-WG weight slices, packed so B-fragments read row-major contiguous K.
  // Layer1 row (q*16+j) = gate q of unit (16*w + j); cols 0..63 = Wxh1, 64..319 = Whh1.
  __shared__ unsigned short sW1[64 * 320];          // 40 KB
  __shared__ unsigned short sW2[64 * 512];          // 64 KB
  __shared__ float sBias[128];                      // [layer][gate*16+j]
  __shared__ unsigned short sX[2][BATCH * IN_DIM];  // 2 x 16 KB TDM double buffer

  const int w      = blockIdx.x;             // owns hidden units [16w, 16w+16)
  const int tid    = threadIdx.x;
  const int lane   = tid & 31;
  const int wave   = tid >> 5;               // 0..7 -> M tile
  const int m0     = wave * 16;
  const int jl     = lane & 15;
  const int jcol   = w * 16 + jl;
  const int rowoff = (lane >> 4) << 3;       // C/D: VGPR e -> row e (+8 upper lanes)

#if HAVE_TDM
  // Kick the t=0 x tile DMA before weight staging; it lands under TENSORcnt.
  if (wave == 0)
    tdm_load_x_tile(xt, (unsigned)(uintptr_t)&sX[0][0]);
#endif

  for (int idx = tid; idx < 64 * 320; idx += NTHREADS) {
    int row = idx / 320, k = idx - row * 320;
    int q = row >> 4, j = row & 15;
    int g = q * HID + w * 16 + j;
    float v = (k < IN_DIM) ? Wxh1[g * IN_DIM + k] : Whh1[g * HID + (k - IN_DIM)];
    sW1[idx] = f2bf(v);
  }
  for (int idx = tid; idx < 64 * 512; idx += NTHREADS) {
    int row = idx >> 9, k = idx & 511;
    int q = row >> 4, j = row & 15;
    int g = q * HID + w * 16 + j;
    float v = (k < HID) ? Wxh2[g * HID + k] : Whh2[g * HID + (k - HID)];
    sW2[idx] = f2bf(v);
  }
  if (tid < 128) {
    int L = tid >> 6, r = tid & 63;
    int q = r >> 4, j = r & 15;
    int g = q * HID + w * 16 + j;
    sBias[tid] = (L == 0) ? (bxh1[g] + bhh1[g]) : (bxh2[g] + bhh2[g]);
  }
#if HAVE_TDM
  if (wave == 0) tdm_wait_all();             // sX[0] complete before barrier
#endif
  __syncthreads();

  // Cell state lives in VGPRs for the entire scan.
  v8f c1 = vzero(), m1 = vzero(), n1 = vzero();
  v8f c2 = vzero(), m2 = vzero(), n2 = vzero();
  v8f h2reg = vzero();
  unsigned int epoch = 0;

  for (int t = 0; t < TLEN; ++t) {
    const unsigned short* h1rd = (t & 1) ? h1a : h1b;   // h1(t-1)
    unsigned short*       h1wr = (t & 1) ? h1b : h1a;   // h1(t)
    const unsigned short* h2rd = (t & 1) ? h2a : h2b;   // h2(t-1)
    unsigned short*       h2wr = (t & 1) ? h2b : h2a;   // h2(t)

#if HAVE_TDM
    // Overlap next timestep's x tile DMA with this step's WMMA work.
    if (wave == 0 && t + 1 < TLEN)
      tdm_load_x_tile(xt + (size_t)(t + 1) * (BATCH * IN_DIM),
                      (unsigned)(uintptr_t)&sX[(t + 1) & 1][0]);
#endif

    // ---------------- layer 1: gates = [x_t | h1] * [Wxh1 | Whh1]^T + bias ----
    v8f acc0, acc1, acc2, acc3;
    {
      float g0 = sBias[jl], g1 = sBias[16 + jl], g2 = sBias[32 + jl], g3 = sBias[48 + jl];
#pragma unroll
      for (int e = 0; e < 8; ++e) { acc0[e] = g0; acc1[e] = g1; acc2[e] = g2; acc3[e] = g3; }
    }
#if HAVE_TDM
    const unsigned short* xrow = &sX[t & 1][0] + m0 * IN_DIM;
#else
    const unsigned short* xrow = xt + (size_t)t * (BATCH * IN_DIM) + m0 * IN_DIM;
#endif
#pragma unroll
    for (int kt = 0; kt < 10; ++kt) {
      v16bf a = (kt < 2) ? load_frag_a(xrow + kt * 32, IN_DIM, lane)
                         : load_frag_a(h1rd + m0 * HID + (kt - 2) * 32, HID, lane);
      v16bf bI = load_frag_b(sW1 +  0 * 320 + kt * 32, 320, lane);
      v16bf bF = load_frag_b(sW1 + 16 * 320 + kt * 32, 320, lane);
      v16bf bZ = load_frag_b(sW1 + 32 * 320 + kt * 32, 320, lane);
      v16bf bO = load_frag_b(sW1 + 48 * 320 + kt * 32, 320, lane);
      acc0 = __builtin_amdgcn_wmma_f32_16x16x32_bf16(false, a, false, bI, (short)0, acc0, false, false);
      acc1 = __builtin_amdgcn_wmma_f32_16x16x32_bf16(false, a, false, bF, (short)0, acc1, false, false);
      acc2 = __builtin_amdgcn_wmma_f32_16x16x32_bf16(false, a, false, bZ, (short)0, acc2, false, false);
      acc3 = __builtin_amdgcn_wmma_f32_16x16x32_bf16(false, a, false, bO, (short)0, acc3, false, false);
    }
    v8f h1reg;
#pragma unroll
    for (int e = 0; e < 8; ++e) {
      float ig = acc0[e], fg = acc1[e];
      float zt = tanh_fast(acc2[e]);
      float ot = sigmoid_fast(acc3[e]);
      float mt = fmaxf(fg + m1[e], ig);
      float it = __expf(ig - mt);
      float ft = __expf(fg + m1[e] - mt);
      float ct = ft * c1[e] + it * zt;
      float nt = ft * n1[e] + it;
      c1[e] = ct; n1[e] = nt; m1[e] = mt;
      h1reg[e] = ot * (ct / nt);
    }
#pragma unroll
    for (int e = 0; e < 8; ++e)
      h1wr[(m0 + rowoff + e) * HID + jcol] = f2bf(h1reg[e]);

    grid_sync(cnt, epoch);   // h1(t) visible device-wide

    // ---------------- layer 2: gates = [h1(t) | h2(t-1)] * [Wxh2 | Whh2]^T ----
    {
      float g0 = sBias[64 + jl], g1 = sBias[80 + jl], g2 = sBias[96 + jl], g3 = sBias[112 + jl];
#pragma unroll
      for (int e = 0; e < 8; ++e) { acc0[e] = g0; acc1[e] = g1; acc2[e] = g2; acc3[e] = g3; }
    }
#pragma unroll
    for (int kt = 0; kt < 16; ++kt) {
      v16bf a = (kt < 8) ? load_frag_a(h1wr + m0 * HID + kt * 32, HID, lane)
                         : load_frag_a(h2rd + m0 * HID + (kt - 8) * 32, HID, lane);
      v16bf bI = load_frag_b(sW2 + (0 * 16) * 512 + kt * 32, 512, lane);
      v16bf bF = load_frag_b(sW2 + (1 * 16) * 512 + kt * 32, 512, lane);
      v16bf bZ = load_frag_b(sW2 + (2 * 16) * 512 + kt * 32, 512, lane);
      v16bf bO = load_frag_b(sW2 + (3 * 16) * 512 + kt * 32, 512, lane);
      acc0 = __builtin_amdgcn_wmma_f32_16x16x32_bf16(false, a, false, bI, (short)0, acc0, false, false);
      acc1 = __builtin_amdgcn_wmma_f32_16x16x32_bf16(false, a, false, bF, (short)0, acc1, false, false);
      acc2 = __builtin_amdgcn_wmma_f32_16x16x32_bf16(false, a, false, bZ, (short)0, acc2, false, false);
      acc3 = __builtin_amdgcn_wmma_f32_16x16x32_bf16(false, a, false, bO, (short)0, acc3, false, false);
    }
#pragma unroll
    for (int e = 0; e < 8; ++e) {
      float ig = acc0[e], fg = acc1[e];
      float zt = tanh_fast(acc2[e]);
      float ot = sigmoid_fast(acc3[e]);
      float mt = fmaxf(fg + m2[e], ig);
      float it = __expf(ig - mt);
      float ft = __expf(fg + m2[e] - mt);
      float ct = ft * c2[e] + it * zt;
      float nt = ft * n2[e] + it;
      c2[e] = ct; n2[e] = nt; m2[e] = mt;
      h2reg[e] = ot * (ct / nt);
    }
#pragma unroll
    for (int e = 0; e < 8; ++e)
      h2wr[(m0 + rowoff + e) * HID + jcol] = f2bf(h2reg[e]);

#if HAVE_TDM
    if (wave == 0) tdm_wait_all();   // x(t+1) resident before the step barrier
#endif
    grid_sync(cnt, epoch);           // h2(t) visible; sX[(t+1)&1] published WG-wide
  }

  // publish final h2 in fp32 for the head MLP
#pragma unroll
  for (int e = 0; e < 8; ++e)
    h2f[(m0 + rowoff + e) * HID + jcol] = h2reg[e];

  grid_sync(cnt, epoch);

  // ---- head MLP on block 0: [B,256]->relu[128]->relu[64]->[1] ----
  if (w == 0 && tid < BATCH) {
    int b = tid;
    float a1[128];
    for (int j = 0; j < 128; ++j) {
      float s = b1[j];
      for (int k = 0; k < 256; ++k) s += h2f[b * 256 + k] * W1[j * 256 + k];
      a1[j] = fmaxf(s, 0.f);
    }
    float a2[64];
    for (int j = 0; j < 64; ++j) {
      float s = b2[j];
      for (int k = 0; k < 128; ++k) s += a1[k] * W2[j * 128 + k];
      a2[j] = fmaxf(s, 0.f);
    }
    float s = b3[0];
    for (int k = 0; k < 64; ++k) s += a2[k] * W3[k];
    out[b] = s;
  }
}

extern "C" void kernel_launch(void* const* d_in, const int* in_sizes, int n_in,
                              void* d_out, int out_size, void* d_ws, size_t ws_size,
                              hipStream_t stream) {
  (void)in_sizes; (void)n_in; (void)out_size; (void)ws_size;
  const float* x    = (const float*)d_in[0];
  const float* Wxh1 = (const float*)d_in[1];
  const float* bxh1 = (const float*)d_in[2];
  const float* Whh1 = (const float*)d_in[3];
  const float* bhh1 = (const float*)d_in[4];
  const float* Wxh2 = (const float*)d_in[5];
  const float* bxh2 = (const float*)d_in[6];
  const float* Whh2 = (const float*)d_in[7];
  const float* bhh2 = (const float*)d_in[8];
  const float* W1   = (const float*)d_in[9];
  const float* b1   = (const float*)d_in[10];
  const float* W2   = (const float*)d_in[11];
  const float* b2   = (const float*)d_in[12];
  const float* W3   = (const float*)d_in[13];
  const float* b3   = (const float*)d_in[14];
  float* out = (float*)d_out;

  char* ws = (char*)d_ws;
  unsigned short* xt = (unsigned short*)ws;                      // 8 MB [T][B][IN] bf16
  char* hb = ws + (size_t)TLEN * BATCH * IN_DIM * 2;
  unsigned short* h1a = (unsigned short*)(hb + 0 * 65536);
  unsigned short* h1b = (unsigned short*)(hb + 1 * 65536);
  unsigned short* h2a = (unsigned short*)(hb + 2 * 65536);
  unsigned short* h2b = (unsigned short*)(hb + 3 * 65536);
  float*          h2f = (float*)(hb + 4 * 65536);                // 128 KB fp32
  unsigned int*   cnt = (unsigned int*)(hb + 4 * 65536 + 131072);

  // zero h state (both parities), h2f, and the barrier counter (graph-capturable)
  hipMemsetAsync(hb, 0, 4 * 65536 + 131072 + 256, stream);

  int nconv = BATCH * TLEN * IN_DIM;
  xt_kernel<<<(nconv + 255) / 256, 256, 0, stream>>>(x, xt);

  slstm_persistent<<<NWG, NTHREADS, 0, stream>>>(
      xt, h1a, h1b, h2a, h2b, h2f,
      Wxh1, bxh1, Whh1, bhh1, Wxh2, bxh2, Whh2, bhh2,
      W1, b1, W2, b2, W3, b3, out, cnt);
}
